// DGCRNModel_35837207118648
// MI455X (gfx1250) — matrix operand order repaired and checked by reading, not compiled, not run on hardware
//
#include <hip/hip_runtime.h>
#include <hip/hip_bf16.h>

// ---------------- problem constants ----------------
#define NN    1024      // nodes
#define BB    32        // batch
#define TIN   12
#define TOUT  12
#define EMBD  10
#define HIDD  64
#define EPSF  1e-6f
#define FP0   96        // padded features layer0 (65 used)
#define FP1   128       // padded features layer1 (128 used)

// Use the Tensor Data Mover for A-tile staging (inline asm). Set to 0 to fall
// back to per-lane GLOBAL_LOAD_ASYNC_TO_LDS (builtin, __has_builtin-guarded)
// or plain VGPR copies.
#define USE_TDM 1

// ---------------- WMMA types ----------------
typedef __bf16 v16bf __attribute__((ext_vector_type(16)));
typedef float  v8f   __attribute__((ext_vector_type(8)));
typedef unsigned int u32x4 __attribute__((ext_vector_type(4)));
typedef unsigned int u32x8 __attribute__((ext_vector_type(8)));

struct __align__(16) U4 { unsigned x, y, z, w; };
union Frag { v16bf v; U4 q[2]; };

__device__ __forceinline__ unsigned short f2bf(float f) {
  union { float f; unsigned u; } c; c.f = f;
  unsigned r = c.u + 0x7FFFu + ((c.u >> 16) & 1u);
  return (unsigned short)(r >> 16);
}
__device__ __forceinline__ float sigm(float x) { return 1.f / (1.f + __expf(-x)); }

// LDS byte offset of a generic pointer that points into LDS
__device__ __forceinline__ unsigned lds_addr_of(void* p) {
  return (unsigned)(unsigned long long)(__attribute__((address_space(3))) char*)(char*)p;
}

#if !USE_TDM
#if __has_builtin(__builtin_amdgcn_global_load_async_to_lds_b128) && \
    __has_builtin(__builtin_amdgcn_s_wait_asynccnt)
#define HAS_ASYNC 1
#else
#define HAS_ASYNC 0
#endif
__device__ __forceinline__ void async_copy16(const void* g, void* l) {
#if HAS_ASYNC
  __builtin_amdgcn_global_load_async_to_lds_b128(
      (__attribute__((address_space(1))) const void*)g,
      (__attribute__((address_space(3))) void*)l, 0, 0);
#else
  *(U4*)l = *(const U4*)g;
#endif
}
__device__ __forceinline__ void async_wait0() {
#if HAS_ASYNC
  __builtin_amdgcn_s_wait_asynccnt(0);
#endif
}
#endif  // !USE_TDM

#if USE_TDM
// Issue a TDM 2D tile copy: rows x cols (bf16 elements) from global (row
// stride = stride_elems) into contiguous LDS at lds_byte_off.
// D# per CDNA5 ISA (group0 128b + group1 256b).
__device__ __forceinline__ void tdm_load_tile(
    unsigned lds_byte_off, const unsigned short* gsrc,
    unsigned rows, unsigned cols, unsigned stride_elems)
{
  unsigned long long ga = (unsigned long long)gsrc;
  u32x4 g0;
  g0[0] = 1u;                                        // count=1 (valid), user mode
  g0[1] = lds_byte_off;                              // lds_addr
  g0[2] = (unsigned)(ga & 0xffffffffu);              // global_addr[31:0]
  g0[3] = (unsigned)((ga >> 32) & 0x01ffffffu)       // global_addr[56:32]
        | (2u << 30);                                // type=2 ("image")
  u32x8 g1;
  unsigned td0 = stride_elems;                       // tensor_dim0 (>= tile cols)
  unsigned td1 = rows;                               // tensor_dim1
  g1[0] = (1u << 16);                                // wg_mask=0, data_size=1 (2B)
  g1[1] = (td0 & 0xffffu) << 16;                     // atomic_addr=0 | tensor_dim0.lo
  g1[2] = (td0 >> 16) | ((td1 & 0xffffu) << 16);     // tensor_dim0.hi | tensor_dim1.lo
  g1[3] = (td1 >> 16) | (cols << 16);                // tensor_dim1.hi | tile_dim0
  g1[4] = rows & 0xffffu;                            // tile_dim1 | tile_dim2=0
  g1[5] = stride_elems;                              // tensor_dim0_stride[31:0]
  g1[6] = 0u;                                        // stride.hi | dim1_stride.lo
  g1[7] = 0u;
  asm volatile("tensor_load_to_lds %0, %1" :: "s"(g0), "s"(g1) : "memory");
}
#endif  // USE_TDM

// =====================================================================
// Generic bf16 WMMA GEMM:  C[M x NC] = A[M x K] * B[K x NC] (+bias)
// Block tile 256(M) x 32(N), K-chunk 32, double-buffered LDS.
// 8 waves; wave w owns M-tiles {2w,2w+1} x N-tiles {0,1}: 4 WMMAs/chunk.
// A tile staged by the Tensor Data Mover (wave 0 issues, TENSORcnt wait),
// B tile staged sync + transposed. Requires M%256==0, NC%32==0, K%32==0.
// =====================================================================
#define GBM 256
#define GBN 32
#define GKB 32

__global__ __launch_bounds__(256) void gemm_bf16_wmma(
    const unsigned short* __restrict__ A, int lda, long long aBatch,
    const unsigned short* __restrict__ Bm, int ldb, long long bBatch,
    void* __restrict__ Cv, int ldc, long long cBatch,
    const float* __restrict__ bias, int K, int storeBf16)
{
  __shared__ __align__(16) unsigned short lA[2][GBM][GKB];   // 2 x 16 KB
  __shared__ __align__(16) unsigned short lBt[2][GBN][GKB];  // 2 x 2 KB ([col][k])

  const int tid  = threadIdx.x;
  const int wv   = tid >> 5;        // 0..7
  const int lane = tid & 31;
  const int half = lane >> 4;       // 0/1
  const int lc   = lane & 15;

  const long long m0 = (long long)blockIdx.x * GBM;
  const long long n0 = (long long)blockIdx.y * GBN;
  const unsigned short* Ab = A  + (long long)blockIdx.z * aBatch;
  const unsigned short* Bp = Bm + (long long)blockIdx.z * bBatch;

  v8f acc[2][2];
  #pragma unroll
  for (int i = 0; i < 2; ++i)
    #pragma unroll
    for (int j = 0; j < 2; ++j)
      #pragma unroll
      for (int e = 0; e < 8; ++e) acc[i][j][e] = 0.f;

  const int brr = tid >> 3;         // 0..31 : B tile k row
  const int bcc = (tid & 7) << 2;   // 0..28 : B tile col chunk (4 elems)

  unsigned long long bReg = 0;      // staged B data (4 bf16)

  auto stageA = [&](int k0, int buf) {
#if USE_TDM
    if (wv == 0)
      tdm_load_tile(lds_addr_of(&lA[buf][0][0]),
                    Ab + m0 * (long long)lda + k0, GBM, GKB, (unsigned)lda);
#else
    const unsigned short* src = Ab + (m0 + tid) * (long long)lda + k0;
    #pragma unroll
    for (int c = 0; c < 4; ++c)
      async_copy16(src + c * 8, &lA[buf][tid][c * 8]);
#endif
  };
  auto waitA = [&]() {
#if USE_TDM
    if (wv == 0) __builtin_amdgcn_s_wait_tensorcnt(0);
#else
    async_wait0();
#endif
  };
  auto loadB = [&](int k0) {
    bReg = *(const unsigned long long*)(Bp + (long long)(k0 + brr) * ldb + n0 + bcc);
  };
  auto storeB = [&](int buf) {
    lBt[buf][bcc + 0][brr] = (unsigned short)(bReg);
    lBt[buf][bcc + 1][brr] = (unsigned short)(bReg >> 16);
    lBt[buf][bcc + 2][brr] = (unsigned short)(bReg >> 32);
    lBt[buf][bcc + 3][brr] = (unsigned short)(bReg >> 48);
  };

  // ---- prologue: stage chunk 0 into buffer 0 ----
  stageA(0, 0);
  loadB(0);
  waitA();
  storeB(0);
  __syncthreads();

  int p = 0;
  for (int k0 = 0; k0 < K; k0 += GKB) {
    const bool last = (k0 + GKB >= K);
    if (!last) {            // overlap next-chunk staging with compute
      stageA(k0 + GKB, p ^ 1);
      loadB(k0 + GKB);
    }

    // ---- compute chunk k0 from buffer p ----
    Frag af[2], bfr[2];
    #pragma unroll
    for (int i = 0; i < 2; ++i) {
      const int row = ((wv * 2 + i) << 4) + lc;
      af[i].q[0] = *(const U4*)&lA[p][row][half << 3];
      af[i].q[1] = *(const U4*)&lA[p][row][16 + (half << 3)];
    }
    #pragma unroll
    for (int j = 0; j < 2; ++j) {
      bfr[j].q[0] = *(const U4*)&lBt[p][(j << 4) + lc][half << 3];
      bfr[j].q[1] = *(const U4*)&lBt[p][(j << 4) + lc][16 + (half << 3)];
    }
    #pragma unroll
    for (int i = 0; i < 2; ++i)
      #pragma unroll
      for (int j = 0; j < 2; ++j)
        acc[i][j] = __builtin_amdgcn_wmma_f32_16x16x32_bf16(
            false, af[i].v, false, bfr[j].v, (short)0, acc[i][j], false, false);

    if (!last) {
      waitA();              // this wave's TDM/async ops landed in LDS
      storeB(p ^ 1);
      __syncthreads();      // all waves' staging visible before next chunk
      p ^= 1;
    }
  }

  // ---- epilogue: C/D layout: lane<16 elem r -> (M=r, N=lc); lane>=16 -> (M=8+r)
  #pragma unroll
  for (int i = 0; i < 2; ++i) {
    const long long rowBase = m0 + ((wv * 2 + i) << 4) + (half << 3);
    if (storeBf16) {
      unsigned short* Cp = (unsigned short*)Cv + (long long)blockIdx.z * cBatch;
      #pragma unroll
      for (int j = 0; j < 2; ++j) {
        const long long col = n0 + (j << 4) + lc;
        #pragma unroll
        for (int r = 0; r < 8; ++r)
          Cp[(rowBase + r) * (long long)ldc + col] = f2bf(acc[i][j][r]);
      }
    } else {
      float* Cp = (float*)Cv + (long long)blockIdx.z * cBatch;
      #pragma unroll
      for (int j = 0; j < 2; ++j) {
        const long long col = n0 + (j << 4) + lc;
        const float bv = bias ? bias[col] : 0.f;
        #pragma unroll
        for (int r = 0; r < 8; ++r)
          Cp[(rowBase + r) * (long long)ldc + col] = acc[i][j][r] + bv;
      }
    }
  }
}

// =====================================================================
// Prep kernels
// =====================================================================
__global__ __launch_bounds__(256) void zero_f32(float* p, long long cnt) {
  long long i = (long long)blockIdx.x * 256 + threadIdx.x;
  if (i < cnt) p[i] = 0.f;
}

// A_dyn = relu(emb1 @ emb2^T); row sums + (atomic) col sums. Block = one row.
__global__ __launch_bounds__(256) void adyn_kernel(
    const float* __restrict__ emb1, const float* __restrict__ emb2,
    float* __restrict__ Adyn, float* __restrict__ rowsum, float* __restrict__ colsum)
{
  __shared__ float e1[EMBD];
  __shared__ float red[256];
  const int i = blockIdx.x;
  if (threadIdx.x < EMBD) e1[threadIdx.x] = emb1[i * EMBD + threadIdx.x];
  __syncthreads();
  float s = 0.f;
  for (int j = threadIdx.x; j < NN; j += 256) {
    float d = 0.f;
    #pragma unroll
    for (int e = 0; e < EMBD; ++e) d += e1[e] * emb2[j * EMBD + e];
    d = d > 0.f ? d : 0.f;
    Adyn[(long long)i * NN + j] = d;
    s += d;
    atomicAdd(&colsum[j], d);
  }
  red[threadIdx.x] = s; __syncthreads();
  for (int st = 128; st > 0; st >>= 1) {
    if (threadIdx.x < st) red[threadIdx.x] += red[threadIdx.x + st];
    __syncthreads();
  }
  if (threadIdx.x == 0) rowsum[i] = red[0];
}

// Wf = a*Wff + (1-a)*Adyn_rownorm ; Wb = a*Wbf + (1-a)*AdynT_rownorm  (bf16 out)
__global__ __launch_bounds__(256) void blend_kernel(
    const float* __restrict__ Wff, const float* __restrict__ Wbf,
    const float* __restrict__ Adyn, const float* __restrict__ rowsum,
    const float* __restrict__ colsum, const float* __restrict__ alpha,
    unsigned short* __restrict__ Wf, unsigned short* __restrict__ Wb)
{
  long long idx = (long long)blockIdx.x * 256 + threadIdx.x;  // over NN*NN
  int i = (int)(idx >> 10), j = (int)(idx & (NN - 1));
  float a = sigm(alpha[0]);
  float wfd = Adyn[idx] / (rowsum[i] + EPSF);
  float wbd = Adyn[(long long)j * NN + i] / (colsum[i] + EPSF);
  Wf[idx] = f2bf(a * Wff[idx] + (1.f - a) * wfd);
  Wb[idx] = f2bf(a * Wbf[idx] + (1.f - a) * wbd);
}

// w (64 x fan) f32 -> w_pad (5*FP x 64) bf16, hop-major with zero padding
__global__ __launch_bounds__(256) void pad_w_kernel(
    const float* __restrict__ w, unsigned short* __restrict__ wp,
    int inTot, int FP, int fan)
{
  int idx = blockIdx.x * 256 + threadIdx.x;  // over 5*FP*64
  int p = idx >> 6, o = idx & 63;
  int s = p / FP, f = p - s * FP;
  float v = (f < inTot) ? w[o * fan + s * inTot + f] : 0.f;
  wp[idx] = f2bf(v);
}

// =====================================================================
// Cell elementwise kernels
// =====================================================================
// hops slot0 = [x_l (inFeat) | h (64) | zero-pad] as bf16
__global__ __launch_bounds__(256) void build_xh(
    const float* __restrict__ xsrc, long long xbs, int inFeat,
    const float* __restrict__ h, unsigned short* __restrict__ hops,
    int FP, int FPt)
{
  long long idx = (long long)blockIdx.x * 256 + threadIdx.x;  // over BB*NN*FP
  long long bn = idx / FP;
  int f = (int)(idx - bn * FP);
  int b = (int)(bn >> 10), n = (int)(bn & (NN - 1));
  float v;
  if (f < inFeat)               v = xsrc[(long long)b * xbs + (long long)n * inFeat + f];
  else if (f < inFeat + HIDD)   v = h[bn * HIDD + (f - inFeat)];
  else                          v = 0.f;
  hops[bn * FPt + f] = f2bf(v);
}

// r=sig(rlin), z=sig(zlin); keep z; overwrite slot0 h-part with r*h (bf16)
__global__ __launch_bounds__(256) void rz_kernel(
    const float* __restrict__ rlin, const float* __restrict__ zlin,
    const float* __restrict__ h, float* __restrict__ zb,
    unsigned short* __restrict__ hops, int inFeat, int FPt)
{
  long long idx = (long long)blockIdx.x * 256 + threadIdx.x;  // over BB*NN*64
  long long bn = idx >> 6;
  int f = (int)(idx & 63);
  float r = sigm(rlin[idx]);
  float z = sigm(zlin[idx]);
  zb[idx] = z;
  hops[bn * FPt + inFeat + f] = f2bf(r * h[idx]);
}

// h = (1-z)*h + z*tanh(nlin)
__global__ __launch_bounds__(256) void update_kernel(
    const float* __restrict__ nlin, const float* __restrict__ zb,
    float* __restrict__ h)
{
  long long idx = (long long)blockIdx.x * 256 + threadIdx.x;
  float z = zb[idx];
  h[idx] = (1.f - z) * h[idx] + z * tanhf(nlin[idx]);
}

// y = h1 @ pw^T + pb ; write ybuf and out[:, t, :]
__global__ __launch_bounds__(256) void proj_kernel(
    const float* __restrict__ h1, const float* __restrict__ pw,
    const float* __restrict__ pb, float* __restrict__ ybuf,
    float* __restrict__ out, int t)
{
  int idx = blockIdx.x * 256 + threadIdx.x;  // over BB*NN
  int b = idx >> 10, n = idx & (NN - 1);
  float s = pb[0];
  #pragma unroll
  for (int f = 0; f < HIDD; ++f) s += h1[(long long)idx * HIDD + f] * pw[f];
  ybuf[idx] = s;
  out[((long long)b * TOUT + t) * NN + n] = s;
}

// =====================================================================
// Host driver
// =====================================================================
extern "C" void kernel_launch(void* const* d_in, const int* in_sizes, int n_in,
                              void* d_out, int out_size, void* d_ws, size_t ws_size,
                              hipStream_t stream) {
  (void)in_sizes; (void)n_in; (void)out_size; (void)ws_size;

  const float* x     = (const float*)d_in[0];
  const float* Wff   = (const float*)d_in[1];
  const float* Wbf   = (const float*)d_in[2];
  const float* emb1  = (const float*)d_in[3];
  const float* emb2  = (const float*)d_in[4];
  const float* alpha = (const float*)d_in[5];
  const float* cw[2][3] = {
    {(const float*)d_in[6],  (const float*)d_in[8],  (const float*)d_in[10]},
    {(const float*)d_in[12], (const float*)d_in[14], (const float*)d_in[16]}};
  const float* cb[2][3] = {
    {(const float*)d_in[7],  (const float*)d_in[9],  (const float*)d_in[11]},
    {(const float*)d_in[13], (const float*)d_in[15], (const float*)d_in[17]}};
  const float* pw = (const float*)d_in[18];
  const float* pb = (const float*)d_in[19];
  float* out = (float*)d_out;

  // ---- workspace carve-up ----
  char* ws = (char*)d_ws;
  size_t off = 0;
  auto alloc = [&](size_t bytes) -> char* {
    char* p = ws + off;
    off = (off + bytes + 255) & ~(size_t)255;
    return p;
  };
  unsigned short* Wf   = (unsigned short*)alloc((size_t)NN * NN * 2);
  unsigned short* Wb   = (unsigned short*)alloc((size_t)NN * NN * 2);
  float* Adyn          = (float*)alloc((size_t)NN * NN * 4);
  float* rowsum        = (float*)alloc((size_t)NN * 4);
  float* colsum        = (float*)alloc((size_t)NN * 4);
  unsigned short* wpad[2][3];
  for (int l = 0; l < 2; ++l) {
    int FP = l ? FP1 : FP0;
    for (int g = 0; g < 3; ++g)
      wpad[l][g] = (unsigned short*)alloc((size_t)5 * FP * 64 * 2);
  }
  unsigned short* hops[2];
  hops[0] = (unsigned short*)alloc((size_t)BB * NN * 5 * FP0 * 2);
  hops[1] = (unsigned short*)alloc((size_t)BB * NN * 5 * FP1 * 2);
  float* h[2];
  h[0] = (float*)alloc((size_t)BB * NN * HIDD * 4);
  h[1] = (float*)alloc((size_t)BB * NN * HIDD * 4);
  float* g1   = (float*)alloc((size_t)BB * NN * HIDD * 4);
  float* g2   = (float*)alloc((size_t)BB * NN * HIDD * 4);
  float* zbuf = (float*)alloc((size_t)BB * NN * HIDD * 4);
  float* ybuf = (float*)alloc((size_t)BB * NN * 4);

  const long long HN = (long long)BB * NN * HIDD;

  // ---- prep: dynamic adjacency + blend + weight padding + zero state ----
  zero_f32<<<dim3((NN + 255) / 256), 256, 0, stream>>>(colsum, NN);
  adyn_kernel<<<dim3(NN), 256, 0, stream>>>(emb1, emb2, Adyn, rowsum, colsum);
  blend_kernel<<<dim3((NN * NN) / 256), 256, 0, stream>>>(
      Wff, Wbf, Adyn, rowsum, colsum, alpha, Wf, Wb);
  for (int l = 0; l < 2; ++l) {
    int FP = l ? FP1 : FP0;
    int inTot = l ? 128 : 65;
    int fan = 5 * inTot;
    for (int g = 0; g < 3; ++g)
      pad_w_kernel<<<dim3((5 * FP * 64) / 256), 256, 0, stream>>>(
          cw[l][g], wpad[l][g], inTot, FP, fan);
  }
  zero_f32<<<dim3((unsigned)((HN + 255) / 256)), 256, 0, stream>>>(h[0], HN);
  zero_f32<<<dim3((unsigned)((HN + 255) / 256)), 256, 0, stream>>>(h[1], HN);

  // ---- GEMM launcher ----
  auto gemm = [&](const unsigned short* A, int lda, long long aB,
                  const unsigned short* Bm, int ldb, long long bB,
                  void* C, int ldc, long long cB, const float* bias,
                  long long M, int NC, int K, int nb, int storeBf) {
    dim3 grid((unsigned)(M / GBM), (unsigned)(NC / GBN), (unsigned)nb);
    gemm_bf16_wmma<<<grid, 256, 0, stream>>>(A, lda, aB, Bm, ldb, bB,
                                             C, ldc, cB, bias, K, storeBf);
  };

  // ---- one DCGRU cell ----
  auto cell = [&](int l, const float* xsrc, long long xbs, int inFeat) {
    const int FP  = l ? FP1 : FP0;
    const int FPt = 5 * FP;
    unsigned short* H = hops[l];
    float* hh = h[l];
    const long long bStride = (long long)NN * FPt;  // per-batch stride in hops

    build_xh<<<dim3((unsigned)(((long long)BB * NN * FP) / 256)), 256, 0, stream>>>(
        xsrc, xbs, inFeat, hh, H, FP, FPt);

    auto diffusion = [&]() {
      // slot1 = Wf@slot0 ; slot2 = Wf@slot1 ; slot3 = Wb@slot0 ; slot4 = Wb@slot3
      gemm(Wf, NN, 0, H + 0 * FP, FPt, bStride, H + 1 * FP, FPt, bStride,
           nullptr, NN, FP, NN, BB, 1);
      gemm(Wf, NN, 0, H + 1 * FP, FPt, bStride, H + 2 * FP, FPt, bStride,
           nullptr, NN, FP, NN, BB, 1);
      gemm(Wb, NN, 0, H + 0 * FP, FPt, bStride, H + 3 * FP, FPt, bStride,
           nullptr, NN, FP, NN, BB, 1);
      gemm(Wb, NN, 0, H + 3 * FP, FPt, bStride, H + 4 * FP, FPt, bStride,
           nullptr, NN, FP, NN, BB, 1);
    };

    diffusion();
    // r,z gate linears over concat hops: (B*N x 5FP) @ (5FP x 64)
    gemm(H, FPt, 0, wpad[l][0], 64, 0, g1, 64, 0, cb[l][0],
         (long long)BB * NN, 64, FPt, 1, 0);
    gemm(H, FPt, 0, wpad[l][1], 64, 0, g2, 64, 0, cb[l][1],
         (long long)BB * NN, 64, FPt, 1, 0);
    rz_kernel<<<dim3((unsigned)(HN / 256)), 256, 0, stream>>>(
        g1, g2, hh, zbuf, H, inFeat, FPt);
    diffusion();
    gemm(H, FPt, 0, wpad[l][2], 64, 0, g1, 64, 0, cb[l][2],
         (long long)BB * NN, 64, FPt, 1, 0);
    update_kernel<<<dim3((unsigned)(HN / 256)), 256, 0, stream>>>(g1, zbuf, hh);
  };

  // ---- encoder ----
  for (int t = 0; t < TIN; ++t) {
    cell(0, x + (size_t)t * NN, (long long)TIN * NN, 1);
    cell(1, h[0], (long long)NN * HIDD, HIDD);
  }
  // ---- decoder ----
  for (int s = 0; s < TOUT; ++s) {
    const float* din = (s == 0) ? (x + (size_t)(TIN - 1) * NN) : ybuf;
    long long dbs = (s == 0) ? (long long)TIN * NN : (long long)NN;
    cell(0, din, dbs, 1);
    cell(1, h[0], (long long)NN * HIDD, HIDD);
    proj_kernel<<<dim3((BB * NN) / 256), 256, 0, stream>>>(h[1], pw, pb, ybuf, out, s);
  }
}
